// ExReorderingLayer_46591805227608
// MI455X (gfx1250) — matrix operand order repaired and checked by reading, not compiled, not run on hardware
//
#include <hip/hip_runtime.h>

// Problem sizes (fixed by the reference)
constexpr int kS = 1024;   // sequence length
constexpr int kB = 32;     // batch
constexpr int kD = 1024;   // embedding dim

typedef __attribute__((ext_vector_type(16))) __bf16 v16bf;
typedef __attribute__((ext_vector_type(4)))  __bf16 v4bf;
typedef __attribute__((ext_vector_type(8)))  float  v8f;
typedef __attribute__((ext_vector_type(4)))  float  v4f;

// ---------------------------------------------------------------------------
// Kernel 1: per-batch mask stats.
// validf[b][j] = mask ? 0 : 1 ; slen[b] = sum(validf) ; langtok[b] = S - slen
// ---------------------------------------------------------------------------
__global__ void __launch_bounds__(256) mask_stats_kernel(
    const unsigned char* __restrict__ mask,   // (B,S) bool, 1 byte each
    float* __restrict__ validf,               // (B,S)
    float* __restrict__ slenf,                // (B,)
    float* __restrict__ langtokf) {           // (B,)
  __shared__ float red[256];
  const int b = blockIdx.x;
  const int t = threadIdx.x;
  float sum = 0.f;
  for (int j = t; j < kS; j += 256) {
    float v = mask[b * kS + j] ? 0.f : 1.f;
    validf[b * kS + j] = v;
    sum += v;
  }
  red[t] = sum;
  __syncthreads();
  for (int off = 128; off > 0; off >>= 1) {
    if (t < off) red[t] += red[t + off];
    __syncthreads();
  }
  if (t == 0) {
    slenf[b] = red[0];
    langtokf[b] = (float)kS - red[0];
  }
}

// ---------------------------------------------------------------------------
// Kernel 2: one wave32 per (i,b).
//   pred[b][i]   = sigmoid(x[i,b,:]·W + bias) * slen[b]
//   invrow[b][i] = 1 / sum_j validf[b][j] * rcp((pred - (j - langtok))^2 + 1e-3)
// ---------------------------------------------------------------------------
__global__ void __launch_bounds__(256) pred_rowsum_kernel(
    const float* __restrict__ x,        // (S,B,D)
    const float* __restrict__ W,        // (D,)
    const float* __restrict__ bias,     // (1,)
    const float* __restrict__ validf,   // (B,S)
    const float* __restrict__ slenf,    // (B,)
    const float* __restrict__ langtokf, // (B,)
    float* __restrict__ pred,           // (B,S)
    float* __restrict__ invrow) {       // (B,S)
  const int lane = threadIdx.x & 31;
  const int wid  = threadIdx.x >> 5;
  const int gw   = blockIdx.x * 8 + wid;   // 0 .. S*B-1
  const int i = gw >> 5;                   // sequence index
  const int b = gw & 31;                   // batch index

  const float* xr = x + (size_t)(i * kB + b) * kD;
  float acc = 0.f;
  for (int d = lane; d < kD; d += 32) acc += xr[d] * W[d];
  for (int off = 16; off > 0; off >>= 1) acc += __shfl_xor(acc, off, 32);

  const float z = acc + bias[0];
  float p = 1.f / (1.f + __expf(-z));
  p *= slenf[b];

  const float lt = langtokf[b];
  const float* vrow = validf + b * kS;
  float rs = 0.f;
  for (int j = lane; j < kS; j += 32) {
    float dist = p - ((float)j - lt);
    rs += vrow[j] * __builtin_amdgcn_rcpf(dist * dist + 0.001f);
  }
  for (int off = 16; off > 0; off >>= 1) rs += __shfl_xor(rs, off, 32);

  if (lane == 0) {
    pred[b * kS + i]   = p;
    invrow[b * kS + i] = 1.f / rs;
  }
}

// ---------------------------------------------------------------------------
// Kernel 3: batched GEMM  out[i,b,:] = sum_j prob(i,b,j) * pe[j,b,:]
// Block tile M=128, N=128, K-step 32, 8 waves (wave w = M sub-tile).
// A (prob) generated in registers per the 16-bit A 16x32 VGPR layout.
// B staged fp32->bf16 into LDS transposed (row = n, 36-elem stride), with
// double-buffered ping-pong so next-tile global loads overlap the WMMAs.
// Each thread stages a 4x4 (j x n) micro-tile and packs K-runs into b64
// stores: 4x ds_store_b64 per tile (near conflict-free) instead of 16x b16.
// ---------------------------------------------------------------------------
constexpr int LDST  = 36;        // bf16 elements per LDS row (32 used + 4 pad)
constexpr int NKIT  = kS / 32;   // 32 K-iterations

__global__ void __launch_bounds__(256) reorder_gemm_kernel(
    const float* __restrict__ pe,       // (S,B,D)
    const float* __restrict__ validf,   // (B,S)
    const float* __restrict__ langtokf, // (B,)
    const float* __restrict__ pred,     // (B,S)
    const float* __restrict__ invrow,   // (B,S)
    float* __restrict__ out) {          // (S,B,D)
  __shared__ __bf16 lds[2][128 * LDST];   // 2 x 9216 B

  const int tid  = threadIdx.x;
  const int lane = tid & 31;
  const int w    = tid >> 5;          // wave 0..7 == M sub-tile
  const int b    = blockIdx.z;
  const int i0   = blockIdx.x * 128;
  const int n0   = blockIdx.y * 128;

  const int half = lane >> 4;         // 0/1 lane half
  const int l16  = lane & 15;         // m (A) / n (B,C) within 16x16 tile

  const float lt = langtokf[b];
  const int i_lane = i0 + w * 16 + l16;
  const float pv = pred[b * kS + i_lane];
  const float iv = invrow[b * kS + i_lane];
  // dist(e, k0) = (pv + lt - k0 - 8*half) - ce,  ce = (e<8 ? e : e+8)
  const float dbase0 = pv + lt - (float)(8 * half);

  v8f c[8] = {};  // 8 N sub-tiles, fp32 accumulators

  // Cooperative staging: thread owns 4 rows (j) x 4 cols (n)
  const int tn = tid & 31;        // col group
  const int tj = tid >> 5;        // row group
  const int nb = 4 * tn;          // n_local base
  const int jb = 4 * tj;          // j_local base
  const float* vbase = validf + b * kS;
  const float* pebase = pe + (size_t)b * kD + n0 + nb;

  v4f  rv[4];      // staged fp32 values (4 rows x 4 cols)
  float vj[4];     // per-row valid flags

  auto load_tile = [&](int k0) {
    #pragma unroll
    for (int r = 0; r < 4; ++r) {
      vj[r] = vbase[k0 + jb + r];
      rv[r] = *(const v4f*)(pebase + (size_t)(k0 + jb + r) * (kB * kD));
    }
  };
  auto store_tile = [&](__bf16* buf) {
    #pragma unroll
    for (int cc = 0; cc < 4; ++cc) {
      v4bf pk;
      #pragma unroll
      for (int r = 0; r < 4; ++r) pk[r] = (__bf16)(rv[r][cc] * vj[r]);
      *(v4bf*)(buf + (nb + cc) * LDST + jb) = pk;   // 8B-aligned b64 store
    }
  };
  auto compute = [&](const __bf16* buf, float dbase) {
    // A fragment 16x32 bf16 (ISA layout): element e -> K = 8*half + ce
    v16bf a;
    #pragma unroll
    for (int e = 0; e < 16; ++e) {
      const float ce = (float)((e < 8) ? e : e + 8);
      const float dist = dbase - ce;
      a[e] = (__bf16)(iv * __builtin_amdgcn_rcpf(dist * dist + 0.001f));
    }
    #pragma unroll
    for (int u = 0; u < 8; ++u) {
      // B frag 32x16: lane n=l16, K = 8*half + ce (matches A layout)
      const __bf16* p0 = buf + (u * 16 + l16) * LDST + 8 * half;
      v4bf q0 = *(const v4bf*)(p0);
      v4bf q1 = *(const v4bf*)(p0 + 4);
      v4bf q2 = *(const v4bf*)(p0 + 16);
      v4bf q3 = *(const v4bf*)(p0 + 20);
      v16bf bm;
      #pragma unroll
      for (int e = 0; e < 4; ++e) {
        bm[e] = q0[e]; bm[4 + e] = q1[e]; bm[8 + e] = q2[e]; bm[12 + e] = q3[e];
      }
      c[u] = __builtin_amdgcn_wmma_f32_16x16x32_bf16(
          false, a, false, bm, (short)0, c[u], false, false);
    }
  };

  // Prologue: stage tile 0
  load_tile(0);
  store_tile(lds[0]);
  __syncthreads();

  // Ping-pong main loop: globals for it+1 issued before the WMMAs of it,
  // LDS store of it+1 lands in the other buffer, single barrier/iteration.
  for (int it = 0; it < NKIT; ++it) {
    const int cur = it & 1;
    if (it + 1 < NKIT) load_tile((it + 1) * 32);
    compute(lds[cur], dbase0 - (float)(it * 32));
    if (it + 1 < NKIT) store_tile(lds[cur ^ 1]);
    __syncthreads();
  }

  // Epilogue: apply row-validity mask, store fp32.
  // C layout: VGPR r holds M = r + 8*half, N = l16
  #pragma unroll
  for (int r = 0; r < 8; ++r) {
    const int i = i0 + w * 16 + 8 * half + r;
    const float vr = validf[b * kS + i];
    #pragma unroll
    for (int u = 0; u < 8; ++u) {
      const int n = n0 + u * 16 + l16;
      out[(size_t)(i * kB + b) * kD + n] = c[u][r] * vr;
    }
  }
}

// ---------------------------------------------------------------------------
extern "C" void kernel_launch(void* const* d_in, const int* in_sizes, int n_in,
                              void* d_out, int out_size, void* d_ws, size_t ws_size,
                              hipStream_t stream) {
  (void)in_sizes; (void)n_in; (void)out_size; (void)ws_size;
  const float* x            = (const float*)d_in[0];
  // d_in[1] = layer_input : unused by the reference computation
  const float* pe           = (const float*)d_in[2];
  const unsigned char* mask = (const unsigned char*)d_in[3];  // bool array
  const float* W            = (const float*)d_in[4];
  const float* bias         = (const float*)d_in[5];
  float* out = (float*)d_out;

  float* ws       = (float*)d_ws;
  float* validf   = ws;               // B*S   = 32768 floats
  float* slenf    = ws + 32768;       // 32
  float* langtokf = ws + 32800;       // 32
  float* predb    = ws + 32832;       // B*S   = 32768
  float* invrow   = ws + 65600;       // B*S   = 32768  (total ~385 KB)

  mask_stats_kernel<<<kB, 256, 0, stream>>>(mask, validf, slenf, langtokf);
  pred_rowsum_kernel<<<(kS * kB) / 8, 256, 0, stream>>>(
      x, W, bias, validf, slenf, langtokf, predb, invrow);
  dim3 grid(kS / 128, kD / 128, kB);
  reorder_gemm_kernel<<<grid, 256, 0, stream>>>(
      pe, validf, langtokf, predb, invrow, out);
}
